// MultiHeadAttention_22780506538684
// MI455X (gfx1250) — matrix operand order repaired and checked by reading, not compiled, not run on hardware
//
#include <hip/hip_runtime.h>

// MI455X (gfx1250) MHA: 3x bf16-WMMA GEMM projections + flash attention.
// Compute-bound (344 GFLOP vs ~320 MB mandatory HBM traffic @ 23.3 TB/s),
// so all matmuls run on v_wmma_f32_16x16x32_bf16 with fp32 accumulate.
//
// Workspace layout (needs 3*B*S*D*2 = 96 MiB):
//   [0)         qp  bf16 (B*S x D)                  row-major per head slab
//   [+BSD*2)    kp  bf16                            row-major per head slab
//   [+2*BSD*2)  vpT bf16, head-transposed [b,h,hd,s] (B-operand-friendly)

#define B_  4
#define S_  2048
#define D_  2048
#define H_  16
#define HD_ 128
#define BSD ((size_t)B_ * S_ * D_)

typedef unsigned short u16;
typedef unsigned int   u32;
typedef __attribute__((ext_vector_type(16))) __bf16 v16bf;
typedef __attribute__((ext_vector_type(8)))  float  v8f;

union FragBF {
  v16bf v;
  uint4 q[2];
  u16   s[16];
};

__device__ __forceinline__ u32 f2bf_u(float f) {
  u32 u = __float_as_uint(f);
  return (u + 0x7FFFu + ((u >> 16) & 1u)) >> 16;   // round-to-nearest-even
}
__device__ __forceinline__ uint2 pack4bf(float4 v) {
  uint2 r;
  r.x = f2bf_u(v.x) | (f2bf_u(v.y) << 16);
  r.y = f2bf_u(v.z) | (f2bf_u(v.w) << 16);
  return r;
}
__device__ __forceinline__ v8f wmma_bf16(const FragBF& a, const FragBF& b, v8f c) {
  return __builtin_amdgcn_wmma_f32_16x16x32_bf16(false, a.v, false, b.v,
                                                 (short)0, c, false, false);
}
// gfx1250 generic addresses keep the LDS byte offset in the low 32 bits.
__device__ __forceinline__ u32 lds_u32(const void* p) {
  return (u32)(uintptr_t)p;
}

// Cross-lane xor-butterfly reductions over each 16-lane group using
// v_permlane16_b32 (pure VALU; keeps the DS pipe free for WMMA fragments).
// Nibble i of {sel1,sel0} = source lane (i ^ mask) within the 16-lane row.
__device__ __forceinline__ float pl16(float x, u32 s0, u32 s1) {
  int v = __float_as_int(x);
  return __int_as_float(
      __builtin_amdgcn_permlane16(v, v, (int)s0, (int)s1, false, false));
}
__device__ __forceinline__ float xor16_max(float x) {
  x = fmaxf(x, pl16(x, 0x67452301u, 0xEFCDAB89u));  // ^1
  x = fmaxf(x, pl16(x, 0x54761032u, 0xDCFE98BAu));  // ^2
  x = fmaxf(x, pl16(x, 0x32107654u, 0xBA98FEDCu));  // ^4
  x = fmaxf(x, pl16(x, 0xFEDCBA98u, 0x76543210u));  // ^8
  return x;
}
__device__ __forceinline__ float xor16_sum(float x) {
  x += pl16(x, 0x67452301u, 0xEFCDAB89u);
  x += pl16(x, 0x54761032u, 0xDCFE98BAu);
  x += pl16(x, 0x32107654u, 0xBA98FEDCu);
  x += pl16(x, 0xFEDCBA98u, 0x76543210u);
  return x;
}

// ---------------------------------------------------------------------------
// C[m,n] = sum_k X[m,k] * W[n,k] + bias[n]   (torch Linear: x @ W.T + b)
// Block: 128x128 tile of C, 256 threads = 8 waves; wave = 64(M) x 32(N).
// fp32 -> bf16 conversion happens in registers while staging into LDS; the
// next K-tile's global loads are issued before the WMMA work (pipelined).
// VTRANS: scatter the output head-transposed ([b,h,hd,s]) for the V path.
// ---------------------------------------------------------------------------
template <bool VTRANS>
__global__ __launch_bounds__(256)
void gemm_bias_bf16(const float* __restrict__ X, const float* __restrict__ W,
                    const float* __restrict__ bias, u16* __restrict__ C,
                    int M, int N, int K) {
  __shared__ u16 lX[128][40];   // 32 cols + 8 pad (keeps 16B row alignment)
  __shared__ u16 lW[128][40];

  const int tid  = threadIdx.x;
  const int lane = tid & 31;
  const int wave = tid >> 5;
  const int lo = lane & 15, hi = lane >> 4;
  const int waveM = wave & 1;
  const int waveN = wave >> 1;
  const int mBase = blockIdx.y * 128;
  const int nBase = blockIdx.x * 128;

  v8f acc[4][2];
  const v8f z8 = {0.f, 0.f, 0.f, 0.f, 0.f, 0.f, 0.f, 0.f};
#pragma unroll
  for (int mt = 0; mt < 4; ++mt)
#pragma unroll
    for (int nt = 0; nt < 2; ++nt) acc[mt][nt] = z8;

  float4 xr[4], wr[4];
  auto gload = [&](int k0) {
#pragma unroll
    for (int i = 0; i < 4; ++i) {
      int e   = tid + i * 256;
      int row = e >> 3;
      int c4  = (e & 7) << 2;
      xr[i] = *(const float4*)&X[(size_t)(mBase + row) * K + k0 + c4];
      wr[i] = *(const float4*)&W[(size_t)(nBase + row) * K + k0 + c4];
    }
  };
  auto lstore = [&]() {
#pragma unroll
    for (int i = 0; i < 4; ++i) {
      int e   = tid + i * 256;
      int row = e >> 3;
      int c4  = (e & 7) << 2;
      *(uint2*)&lX[row][c4] = pack4bf(xr[i]);
      *(uint2*)&lW[row][c4] = pack4bf(wr[i]);
    }
  };

  gload(0);
  for (int k0 = 0; k0 < K; k0 += 32) {
    __syncthreads();
    lstore();
    __syncthreads();
    if (k0 + 32 < K) gload(k0 + 32);   // overlap next tile with WMMA below

    FragBF a[4], b[2];
#pragma unroll
    for (int mt = 0; mt < 4; ++mt) {
      int r = waveM * 64 + mt * 16 + lo;
      a[mt].q[0] = *(const uint4*)&lX[r][16 * hi];
      a[mt].q[1] = *(const uint4*)&lX[r][16 * hi + 8];
    }
#pragma unroll
    for (int nt = 0; nt < 2; ++nt) {
      int r = waveN * 32 + nt * 16 + lo;
      b[nt].q[0] = *(const uint4*)&lW[r][16 * hi];
      b[nt].q[1] = *(const uint4*)&lW[r][16 * hi + 8];
    }
#pragma unroll
    for (int mt = 0; mt < 4; ++mt)
#pragma unroll
      for (int nt = 0; nt < 2; ++nt)
        acc[mt][nt] = wmma_bf16(a[mt], b[nt], acc[mt][nt]);
  }

#pragma unroll
  for (int mt = 0; mt < 4; ++mt)
#pragma unroll
    for (int nt = 0; nt < 2; ++nt) {
      int n = nBase + waveN * 32 + nt * 16 + lo;
      float bv = bias[n];
#pragma unroll
      for (int r = 0; r < 8; ++r) {
        int m = mBase + waveM * 64 + mt * 16 + hi * 8 + r;
        u16 val = (u16)f2bf_u(acc[mt][nt][r] + bv);
        if (!VTRANS) {
          C[(size_t)m * N + n] = val;
        } else {
          int b_   = m >> 11;                // m / S_
          int r2   = m & (S_ - 1);
          int flat = r2 * D_ + n;            // < 2^23
          int h    = flat >> 18;             // / (S_*HD_)
          int rem  = flat & (S_ * HD_ - 1);
          int s    = rem >> 7;               // / HD_
          int hd   = rem & (HD_ - 1);
          C[(size_t)b_ * S_ * D_ + (size_t)h * S_ * HD_ +
            (size_t)hd * S_ + s] = val;
        }
      }
    }
}

// ---------------------------------------------------------------------------
// Flash attention over contiguous (S x HD) head slabs.
// Grid: (S/128, B*H). Block: 256 threads = 8 waves; wave owns 16 q rows.
// K tile [32][128] and head-transposed V tile [128][32] staged via
// GLOBAL_LOAD_ASYNC_TO_LDS_B128 with LDS double buffering (ASYNCcnt).
// ---------------------------------------------------------------------------
__global__ __launch_bounds__(256)
void flash_attn(const u16* __restrict__ QP, const u16* __restrict__ KP,
                const u16* __restrict__ VPT, const float* __restrict__ mask,
                float* __restrict__ out) {
  __shared__ u16 lK[2][32][136];   // K tile  [krow][hd], padded, 2 buffers
  __shared__ u16 lVt[2][128][40];  // V tile  [hd][krow], padded, 2 buffers
  __shared__ u16 lP[8][16][40];    // per-wave P staging (C-layout->A-layout)

  const int tid = threadIdx.x, lane = tid & 31, wave = tid >> 5;
  const int lo = lane & 15, hi = lane >> 4;
  const size_t slab = (size_t)blockIdx.y * (size_t)S_ * HD_;
  const int qBase = blockIdx.x * 128 + wave * 16;

  const u16* Q  = QP + slab;
  const u16* Kp = KP + slab;
  const u16* Vt = VPT + slab;

  auto stage = [&](int j0, int buf) {
#pragma unroll
    for (int i = 0; i < 2; ++i) {
      int e = tid + i * 256;
      int row = e >> 4, c8 = (e & 15) << 3;
      u32 ka = lds_u32(&lK[buf][row][c8]);
      u32 ko = (u32)(((j0 + row) * HD_ + c8) * sizeof(u16));
      asm volatile("global_load_async_to_lds_b128 %0, %1, %2"
                   :: "v"(ka), "v"(ko), "s"(Kp) : "memory");
      int vr = e >> 2, vc = (e & 3) << 3;
      u32 va = lds_u32(&lVt[buf][vr][vc]);
      u32 vo = (u32)((vr * S_ + j0 + vc) * sizeof(u16));
      asm volatile("global_load_async_to_lds_b128 %0, %1, %2"
                   :: "v"(va), "v"(vo), "s"(Vt) : "memory");
    }
  };

  FragBF qf[4];
  {
    const u16* qrow = Q + (size_t)(qBase + lo) * HD_;
#pragma unroll
    for (int c = 0; c < 4; ++c) {
      qf[c].q[0] = *(const uint4*)(qrow + 32 * c + 16 * hi);
      qf[c].q[1] = *(const uint4*)(qrow + 32 * c + 16 * hi + 8);
    }
  }

  v8f o[8];
  const v8f z8 = {0.f, 0.f, 0.f, 0.f, 0.f, 0.f, 0.f, 0.f};
#pragma unroll
  for (int nt = 0; nt < 8; ++nt) o[nt] = z8;
  float mrow[8], lrow[8];
#pragma unroll
  for (int r = 0; r < 8; ++r) { mrow[r] = -1e30f; lrow[r] = 0.f; }

  const float scale = 0.08838834764831845f;  // 1/sqrt(HD)

  stage(0, 0);
  for (int j0 = 0; j0 < S_; j0 += 32) {
    const int cur = (j0 >> 5) & 1;
    __syncthreads();                 // everyone done reading buf cur^1
    if (j0 + 32 < S_) {
      stage(j0 + 32, cur ^ 1);       // DMA next tile during compute
      asm volatile("s_wait_asynccnt 0x4" ::: "memory");
    } else {
      asm volatile("s_wait_asynccnt 0x0" ::: "memory");
    }
    __syncthreads();                 // all waves' current-tile copies done

    // scores: Sa = k cols j0+lo, Sb = j0+16+lo.
    v8f sa = z8, sb = z8;
#pragma unroll
    for (int c = 0; c < 4; ++c) {
      FragBF b0, b1;
      b0.q[0] = *(const uint4*)&lK[cur][lo][32 * c + 16 * hi];
      b0.q[1] = *(const uint4*)&lK[cur][lo][32 * c + 16 * hi + 8];
      b1.q[0] = *(const uint4*)&lK[cur][16 + lo][32 * c + 16 * hi];
      b1.q[1] = *(const uint4*)&lK[cur][16 + lo][32 * c + 16 * hi + 8];
      sa = wmma_bf16(qf[c], b0, sa);
      sb = wmma_bf16(qf[c], b1, sb);
    }

    // Online softmax over the 32 new columns (VALU permlane reductions).
    float pa[8], pb[8], alpha[8];
    const float* mrowp = mask + (size_t)(qBase + hi * 8) * S_ + j0 + lo;
#pragma unroll
    for (int r = 0; r < 8; ++r) {
      float va = sa[r] * scale + mrowp[(size_t)r * S_];
      float vb = sb[r] * scale + mrowp[(size_t)r * S_ + 16];
      float t = xor16_max(fmaxf(va, vb));
      float mn = fmaxf(mrow[r], t);
      float ea = __expf(va - mn), eb = __expf(vb - mn);
      float rs = xor16_sum(ea + eb);
      float al = __expf(mrow[r] - mn);
      lrow[r] = lrow[r] * al + rs;
      mrow[r] = mn;
      alpha[r] = al;
      pa[r] = ea;
      pb[r] = eb;
    }
#pragma unroll
    for (int nt = 0; nt < 8; ++nt)
#pragma unroll
      for (int r = 0; r < 8; ++r) o[nt][r] *= alpha[r];

    // P: C-layout -> LDS -> A-layout (wave-private region; DS in-order).
#pragma unroll
    for (int r = 0; r < 8; ++r) {
      lP[wave][hi * 8 + r][lo]      = (u16)f2bf_u(pa[r]);
      lP[wave][hi * 8 + r][16 + lo] = (u16)f2bf_u(pb[r]);
    }
    asm volatile("s_wait_dscnt 0" ::: "memory");
    FragBF pf;
    pf.q[0] = *(const uint4*)&lP[wave][lo][16 * hi];
    pf.q[1] = *(const uint4*)&lP[wave][lo][16 * hi + 8];

    // O += P(16x32) @ V(32x128): B frags straight from transposed V tile.
#pragma unroll
    for (int nt = 0; nt < 8; nt += 2) {
      FragBF bv0, bv1;
      bv0.q[0] = *(const uint4*)&lVt[cur][nt * 16 + lo][16 * hi];
      bv0.q[1] = *(const uint4*)&lVt[cur][nt * 16 + lo][16 * hi + 8];
      bv1.q[0] = *(const uint4*)&lVt[cur][(nt + 1) * 16 + lo][16 * hi];
      bv1.q[1] = *(const uint4*)&lVt[cur][(nt + 1) * 16 + lo][16 * hi + 8];
      o[nt]     = wmma_bf16(pf, bv0, o[nt]);
      o[nt + 1] = wmma_bf16(pf, bv1, o[nt + 1]);
    }
  }

#pragma unroll
  for (int r = 0; r < 8; ++r) lrow[r] = 1.0f / lrow[r];
  float* oslab = out + slab;
#pragma unroll
  for (int nt = 0; nt < 8; ++nt)
#pragma unroll
    for (int r = 0; r < 8; ++r)
      oslab[(size_t)(qBase + hi * 8 + r) * HD_ + nt * 16 + lo] =
          o[nt][r] * lrow[r];
}

// ---------------------------------------------------------------------------
extern "C" void kernel_launch(void* const* d_in, const int* in_sizes, int n_in,
                              void* d_out, int out_size, void* d_ws,
                              size_t ws_size, hipStream_t stream) {
  (void)in_sizes; (void)n_in; (void)out_size; (void)ws_size;
  const float* q    = (const float*)d_in[0];
  const float* k    = (const float*)d_in[1];
  const float* v    = (const float*)d_in[2];
  const float* mask = (const float*)d_in[3];
  const float* Wq   = (const float*)d_in[4];
  const float* bq   = (const float*)d_in[5];
  const float* Wk   = (const float*)d_in[6];
  const float* bk   = (const float*)d_in[7];
  const float* Wv   = (const float*)d_in[8];
  const float* bv   = (const float*)d_in[9];
  float* out = (float*)d_out;

  u16* qp  = (u16*)d_ws;
  u16* kp  = qp + BSD;
  u16* vpt = kp + BSD;

  const int M = B_ * S_;  // 8192
  dim3 blk(256);
  dim3 gGemm(D_ / 128, M / 128);       // (16, 64)
  gemm_bias_bf16<false><<<gGemm, blk, 0, stream>>>(q, Wq, bq, qp, M, D_, D_);
  gemm_bias_bf16<false><<<gGemm, blk, 0, stream>>>(k, Wk, bk, kp, M, D_, D_);
  gemm_bias_bf16<true ><<<gGemm, blk, 0, stream>>>(v, Wv, bv, vpt, M, D_, D_);

  dim3 gAttn(S_ / 128, B_ * H_);       // (16, 64)
  flash_attn<<<gAttn, blk, 0, stream>>>(qp, kp, vpt, mask, out);
}
